// Encoder_22660247454494
// MI455X (gfx1250) — compile-verified, matrix operand order checked by reading, not run on hardware
//
#include <hip/hip_runtime.h>
#include <hip/hip_bf16.h>

#define L_SEQ 512
#define B_SZ  64
#define D_DIM 512
#define G3D   1536   // 3*D

typedef __bf16 bf16_t;
typedef __attribute__((ext_vector_type(16))) __bf16 v16bf;
typedef __attribute__((ext_vector_type(8)))  __bf16 bf16x8;
typedef __attribute__((ext_vector_type(8)))  float  v8f;

// ---------------------------------------------------------------------------
// 16x32 bf16 WMMA operand tile load from row-major [rows][512] bf16 matrix
// (fast axis = K). Matches CDNA5 16-bit A/B-matrix VGPR striping:
//   lanes 0-15 : row = lane,    K = {k0..k0+7}, {k0+16..k0+23}
//   lanes 16-31: row = lane-16, K = {k0+8..k0+15}, {k0+24..k0+31}
// Two contiguous 16-byte loads per lane -> clause of global_load_b128.
// ---------------------------------------------------------------------------
static __device__ __forceinline__ v16bf load_tile(const bf16_t* __restrict__ base,
                                                  int ld, int r0, int k0, int lane) {
  const int r  = lane & 15;
  const int ks = (lane >> 4) << 3;          // 0 or 8
  const bf16_t* p = base + (size_t)(r0 + r) * ld + (k0 + ks);
  bf16x8 lo = *(const bf16x8*)(p);
  bf16x8 hi = *(const bf16x8*)(p + 16);
  return __builtin_shufflevector(lo, hi, 0,1,2,3,4,5,6,7,8,9,10,11,12,13,14,15);
}

static __device__ __forceinline__ v8f wmma_bf16(v16bf a, v16bf b, v8f c) {
  return __builtin_amdgcn_wmma_f32_16x16x32_bf16(false, a, false, b, (short)0, c,
                                                 false, false);
}

static __device__ __forceinline__ float sigmoidf_fast(float x) {
  return 1.0f / (1.0f + __expf(-x));
}

// ---------------------------------------------------------------------------
// Weight prep: out[n*K + k] = bf16(in[k*N + n])   (transpose + f32->bf16)
// ---------------------------------------------------------------------------
__global__ void transpose_bf16_kernel(const float* __restrict__ in,
                                      bf16_t* __restrict__ out, int K, int N) {
  size_t total = (size_t)K * N;
  for (size_t i = (size_t)blockIdx.x * blockDim.x + threadIdx.x; i < total;
       i += (size_t)gridDim.x * blockDim.x) {
    size_t n = i / K;
    size_t k = i - n * K;
    out[i] = (bf16_t)in[k * (size_t)N + n];
  }
}

// Embedding gather into bf16 activations X[L*B][D]
__global__ void gather_embed_kernel(const int* __restrict__ xs,
                                    const float* __restrict__ emb,
                                    bf16_t* __restrict__ X) {
  size_t total = (size_t)L_SEQ * B_SZ * D_DIM;
  for (size_t i = (size_t)blockIdx.x * blockDim.x + threadIdx.x; i < total;
       i += (size_t)gridDim.x * blockDim.x) {
    size_t row = i / D_DIM;
    int    d   = (int)(i - row * D_DIM);
    int    tok = xs[row];
    X[i] = (bf16_t)emb[(size_t)tok * D_DIM + d];
  }
}

__global__ void zero_h_kernel(float* __restrict__ zf, bf16_t* __restrict__ zb) {
  int i = blockIdx.x * blockDim.x + threadIdx.x;
  if (i < B_SZ * D_DIM) { zf[i] = 0.0f; zb[i] = (bf16_t)0.0f; }
}

// ---------------------------------------------------------------------------
// C[M,N] = A[M,512] @ Bt[N,512]^T + bias[N]  (bf16 WMMA, f32 accumulate)
// 4 waves per block in a 2x2 arrangement; each wave owns a 32x32 output tile
// (4 accumulators). Per K-chunk: 4 operand tiles -> 4 WMMAs.
// ---------------------------------------------------------------------------
__global__ __launch_bounds__(128) void gemm_bias_kernel(
    const bf16_t* __restrict__ A, const bf16_t* __restrict__ Bt,
    const float* __restrict__ bias, float* __restrict__ C, int M, int N) {
  const int lane = threadIdx.x & 31;
  const int wave = threadIdx.x >> 5;
  const int m0 = blockIdx.x * 64 + (wave & 1) * 32;
  const int n0 = blockIdx.y * 64 + (wave >> 1) * 32;
  if (m0 >= M) return;
  v8f c00 = {}, c01 = {}, c10 = {}, c11 = {};
#pragma unroll 2
  for (int kc = 0; kc < D_DIM; kc += 32) {
    v16bf a0 = load_tile(A,  D_DIM, m0,      kc, lane);
    v16bf a1 = load_tile(A,  D_DIM, m0 + 16, kc, lane);
    v16bf b0 = load_tile(Bt, D_DIM, n0,      kc, lane);
    v16bf b1 = load_tile(Bt, D_DIM, n0 + 16, kc, lane);
    c00 = wmma_bf16(a0, b0, c00);
    c01 = wmma_bf16(a0, b1, c01);
    c10 = wmma_bf16(a1, b0, c10);
    c11 = wmma_bf16(a1, b1, c11);
  }
  const int nn = lane & 15;
  const int mo = (lane >> 4) << 3;
#pragma unroll
  for (int j = 0; j < 8; ++j) {
    const int mA = m0 + j + mo, mB = m0 + 16 + j + mo;
    C[(size_t)mA * N + n0 + nn]      = c00[j] + bias[n0 + nn];
    C[(size_t)mA * N + n0 + 16 + nn] = c01[j] + bias[n0 + 16 + nn];
    C[(size_t)mB * N + n0 + nn]      = c10[j] + bias[n0 + nn];
    C[(size_t)mB * N + n0 + 16 + nn] = c11[j] + bias[n0 + 16 + nn];
  }
}

// ---------------------------------------------------------------------------
// GRU gate epilogue for one 16x16 subtile with accumulators az/ar/ah.
// ---------------------------------------------------------------------------
static __device__ __forceinline__ void gru_epilogue(
    const float* __restrict__ Gt, const float* __restrict__ maskt,
    const float* __restrict__ hprev, float* __restrict__ hout,
    bf16_t* __restrict__ houtb, int b0, int n0, int lane,
    v8f az, v8f ar, v8f ah) {
  const int n  = n0 + (lane & 15);
  const int mo = (lane >> 4) << 3;
#pragma unroll
  for (int j = 0; j < 8; ++j) {
    const int   b  = b0 + j + mo;
    const float gz = Gt[(size_t)b * G3D + n];
    const float gr = Gt[(size_t)b * G3D + D_DIM + n];
    const float gh = Gt[(size_t)b * G3D + 2 * D_DIM + n];
    const float hp = hprev[(size_t)b * D_DIM + n];
    const float z  = sigmoidf_fast(gz + az[j]);
    const float r  = sigmoidf_fast(gr + ar[j]);
    const float hc = tanhf(gh + r * ah[j]);
    float hn = (1.0f - z) * hp + z * hc;
    const float m = maskt[b];
    hn = m * hn + (1.0f - m) * hp;
    hout[(size_t)b * D_DIM + n]  = hn;
    houtb[(size_t)b * D_DIM + n] = (bf16_t)hn;
  }
}

// ---------------------------------------------------------------------------
// One fused GRU recurrence step. Each wave owns a 32x32 tile of h_next and
// accumulates all three gates (z/r/h) for it: 12 WMMA accumulators, and the
// 2 A-tiles (h_prev rows) are reused across the 3 gates -> 8 tile loads per
// 12 WMMAs per K-chunk. Grid: (D/32, B/32) single-wave blocks.
// ---------------------------------------------------------------------------
__global__ __launch_bounds__(32) void gru_step_kernel(
    const float* __restrict__ Gt,       // [B, 1536] x@W + b at this timestep
    const float* __restrict__ maskt,    // [B]
    const float* __restrict__ hprev,    // [B, D] f32
    const bf16_t* __restrict__ hprevb,  // [B, D] bf16
    const bf16_t* __restrict__ Ut,      // [1536, 512] bf16 (U transposed)
    float* __restrict__ hout,           // [B, D] f32
    bf16_t* __restrict__ houtb)         // [B, D] bf16
{
  const int lane = threadIdx.x;
  const int n0 = blockIdx.x * 32;
  const int b0 = blockIdx.y * 32;
  v8f z00 = {}, z01 = {}, z10 = {}, z11 = {};
  v8f r00 = {}, r01 = {}, r10 = {}, r11 = {};
  v8f h00 = {}, h01 = {}, h10 = {}, h11 = {};
#pragma unroll 1
  for (int kc = 0; kc < D_DIM; kc += 32) {
    v16bf a0  = load_tile(hprevb, D_DIM, b0,      kc, lane);
    v16bf a1  = load_tile(hprevb, D_DIM, b0 + 16, kc, lane);
    v16bf bz0 = load_tile(Ut, D_DIM, n0,                  kc, lane);
    v16bf bz1 = load_tile(Ut, D_DIM, n0 + 16,             kc, lane);
    v16bf br0 = load_tile(Ut, D_DIM, D_DIM + n0,          kc, lane);
    v16bf br1 = load_tile(Ut, D_DIM, D_DIM + n0 + 16,     kc, lane);
    v16bf bh0 = load_tile(Ut, D_DIM, 2 * D_DIM + n0,      kc, lane);
    v16bf bh1 = load_tile(Ut, D_DIM, 2 * D_DIM + n0 + 16, kc, lane);
    z00 = wmma_bf16(a0, bz0, z00);  z01 = wmma_bf16(a0, bz1, z01);
    z10 = wmma_bf16(a1, bz0, z10);  z11 = wmma_bf16(a1, bz1, z11);
    r00 = wmma_bf16(a0, br0, r00);  r01 = wmma_bf16(a0, br1, r01);
    r10 = wmma_bf16(a1, br0, r10);  r11 = wmma_bf16(a1, br1, r11);
    h00 = wmma_bf16(a0, bh0, h00);  h01 = wmma_bf16(a0, bh1, h01);
    h10 = wmma_bf16(a1, bh0, h10);  h11 = wmma_bf16(a1, bh1, h11);
  }
  gru_epilogue(Gt, maskt, hprev, hout, houtb, b0,      n0,      lane, z00, r00, h00);
  gru_epilogue(Gt, maskt, hprev, hout, houtb, b0,      n0 + 16, lane, z01, r01, h01);
  gru_epilogue(Gt, maskt, hprev, hout, houtb, b0 + 16, n0,      lane, z10, r10, h10);
  gru_epilogue(Gt, maskt, hprev, hout, houtb, b0 + 16, n0 + 16, lane, z11, r11, h11);
}

// s0 = tanh(h_back @ Ws + bs) : (64x512)*(512x512), 32x32 tile per wave
__global__ __launch_bounds__(32) void s0_kernel(
    const bf16_t* __restrict__ A,   // [B, 512] bf16 h_back
    const bf16_t* __restrict__ Bt,  // [512, 512] bf16 (Ws transposed)
    const float* __restrict__ bias, // [512]
    float* __restrict__ out)        // [B, 512]
{
  const int lane = threadIdx.x;
  const int n0 = blockIdx.x * 32;
  const int m0 = blockIdx.y * 32;
  v8f c00 = {}, c01 = {}, c10 = {}, c11 = {};
#pragma unroll 2
  for (int kc = 0; kc < D_DIM; kc += 32) {
    v16bf a0 = load_tile(A,  D_DIM, m0,      kc, lane);
    v16bf a1 = load_tile(A,  D_DIM, m0 + 16, kc, lane);
    v16bf b0 = load_tile(Bt, D_DIM, n0,      kc, lane);
    v16bf b1 = load_tile(Bt, D_DIM, n0 + 16, kc, lane);
    c00 = wmma_bf16(a0, b0, c00);
    c01 = wmma_bf16(a0, b1, c01);
    c10 = wmma_bf16(a1, b0, c10);
    c11 = wmma_bf16(a1, b1, c11);
  }
  const int nn = lane & 15;
  const int mo = (lane >> 4) << 3;
#pragma unroll
  for (int j = 0; j < 8; ++j) {
    const int mA = m0 + j + mo, mB = m0 + 16 + j + mo;
    out[(size_t)mA * D_DIM + n0 + nn]      = tanhf(c00[j] + bias[n0 + nn]);
    out[(size_t)mA * D_DIM + n0 + 16 + nn] = tanhf(c01[j] + bias[n0 + 16 + nn]);
    out[(size_t)mB * D_DIM + n0 + nn]      = tanhf(c10[j] + bias[n0 + nn]);
    out[(size_t)mB * D_DIM + n0 + 16 + nn] = tanhf(c11[j] + bias[n0 + 16 + nn]);
  }
}

// ---------------------------------------------------------------------------
extern "C" void kernel_launch(void* const* d_in, const int* in_sizes, int n_in,
                              void* d_out, int out_size, void* d_ws, size_t ws_size,
                              hipStream_t stream) {
  (void)in_sizes; (void)n_in; (void)out_size; (void)ws_size;
  const int*   xs   = (const int*)d_in[0];
  const float* mask = (const float*)d_in[1];
  const float* emb  = (const float*)d_in[2];
  const float* Wf   = (const float*)d_in[3];
  const float* Uf   = (const float*)d_in[4];
  const float* bf   = (const float*)d_in[5];
  const float* Wb   = (const float*)d_in[6];
  const float* Ub   = (const float*)d_in[7];
  const float* bb   = (const float*)d_in[8];
  const float* Ws   = (const float*)d_in[9];
  const float* bs   = (const float*)d_in[10];

  float* enc = (float*)d_out;                                  // [L,B,D]
  float* s0  = (float*)d_out + (size_t)L_SEQ * B_SZ * D_DIM;   // [B,D]

  // -------- workspace carve-out --------
  uint8_t* ws = (uint8_t*)d_ws;
  size_t off = 0;
  auto alloc = [&](size_t bytes) -> void* {
    void* p = ws + off;
    off = (off + bytes + 255) & ~(size_t)255;
    return p;
  };
  const size_t WDSZ = (size_t)G3D * D_DIM;     // 512*1536 elements
  const size_t LBD  = (size_t)L_SEQ * B_SZ * D_DIM;
  const size_t BD   = (size_t)B_SZ * D_DIM;

  bf16_t* Wft  = (bf16_t*)alloc(WDSZ * 2);           // [1536][512]
  bf16_t* Uft  = (bf16_t*)alloc(WDSZ * 2);
  bf16_t* Wbt  = (bf16_t*)alloc(WDSZ * 2);
  bf16_t* Ubt  = (bf16_t*)alloc(WDSZ * 2);
  bf16_t* Wst  = (bf16_t*)alloc((size_t)D_DIM * D_DIM * 2);
  bf16_t* Xbf  = (bf16_t*)alloc(LBD * 2);            // embedded inputs, bf16
  float*  rightf = (float*)alloc(LBD * 4);           // forward h sequence f32
  bf16_t* rightb = (bf16_t*)alloc(LBD * 2);          // forward h sequence bf16
  float*  Gf   = (float*)alloc((size_t)L_SEQ * B_SZ * G3D * 4);
  float*  Gb   = (float*)alloc((size_t)L_SEQ * B_SZ * G3D * 4);
  float*  zf   = (float*)alloc(BD * 4);              // zero h0 f32
  bf16_t* zb   = (bf16_t*)alloc(BD * 2);             // zero h0 bf16
  bf16_t* hb   = (bf16_t*)alloc(2 * BD * 2);         // backward h bf16 ping-pong

  // -------- weight prep (transpose + bf16) --------
  transpose_bf16_kernel<<<1024, 256, 0, stream>>>(Wf, Wft, D_DIM, G3D);
  transpose_bf16_kernel<<<1024, 256, 0, stream>>>(Uf, Uft, D_DIM, G3D);
  transpose_bf16_kernel<<<1024, 256, 0, stream>>>(Wb, Wbt, D_DIM, G3D);
  transpose_bf16_kernel<<<1024, 256, 0, stream>>>(Ub, Ubt, D_DIM, G3D);
  transpose_bf16_kernel<<<512, 256, 0, stream>>>(Ws, Wst, D_DIM, D_DIM);

  // -------- embedding gather --------
  gather_embed_kernel<<<4096, 256, 0, stream>>>(xs, emb, Xbf);

  // -------- Gf = X @ Wf + bf  (M=32768, N=1536) --------
  {
    dim3 grid(L_SEQ * B_SZ / 64, G3D / 64);
    gemm_bias_kernel<<<grid, 128, 0, stream>>>(Xbf, Wft, bf, Gf,
                                               L_SEQ * B_SZ, G3D);
  }

  // -------- forward GRU scan --------
  zero_h_kernel<<<(B_SZ * D_DIM + 255) / 256, 256, 0, stream>>>(zf, zb);
  {
    dim3 grid(D_DIM / 32, B_SZ / 32);
    for (int t = 0; t < L_SEQ; ++t) {
      const float*  hp  = (t == 0) ? zf : rightf + (size_t)(t - 1) * BD;
      const bf16_t* hpb = (t == 0) ? zb : rightb + (size_t)(t - 1) * BD;
      gru_step_kernel<<<grid, 32, 0, stream>>>(
          Gf + (size_t)t * B_SZ * G3D, mask + (size_t)t * B_SZ,
          hp, hpb, Uft,
          rightf + (size_t)t * BD, rightb + (size_t)t * BD);
    }
  }

  // -------- Gb = right @ Wb + bb (forward-time order) --------
  {
    dim3 grid(L_SEQ * B_SZ / 64, G3D / 64);
    gemm_bias_kernel<<<grid, 128, 0, stream>>>(rightb, Wbt, bb, Gb,
                                               L_SEQ * B_SZ, G3D);
  }

  // -------- backward GRU scan (writes enc[t] = left_rev[L-1-t] directly) ----
  zero_h_kernel<<<(B_SZ * D_DIM + 255) / 256, 256, 0, stream>>>(zf, zb);
  {
    dim3 grid(D_DIM / 32, B_SZ / 32);
    for (int s = 0; s < L_SEQ; ++s) {
      const int t = L_SEQ - 1 - s;  // forward-time index of this backward step
      const float*  hp  = (s == 0) ? zf : enc + (size_t)(t + 1) * BD;
      const bf16_t* hpb = (s == 0) ? zb : hb + (size_t)((s - 1) & 1) * BD;
      gru_step_kernel<<<grid, 32, 0, stream>>>(
          Gb + (size_t)t * B_SZ * G3D, mask + (size_t)t * B_SZ,
          hp, hpb, Ubt,
          enc + (size_t)t * BD, hb + (size_t)(s & 1) * BD);
    }
  }

  // -------- s0 = tanh(h_back @ Ws + bs); final backward write was hb[1] -----
  {
    dim3 grid(D_DIM / 32, B_SZ / 32);
    s0_kernel<<<grid, 32, 0, stream>>>(hb + ((L_SEQ - 1) & 1) * BD, Wst, bs, s0);
  }
}
